// Self_Attention_44779329028527
// MI455X (gfx1250) — compile-verified, matrix operand order checked by reading, not run on hardware
//
#include <hip/hip_runtime.h>
#include <math.h>

typedef __attribute__((ext_vector_type(16))) __bf16 v16bf;
typedef __attribute__((ext_vector_type(8)))  float  v8f;

constexpr int B = 8, S = 2048, D = 1024;
constexpr float SCALE  = 0.125f;                  // 1/sqrt(64)
constexpr float LOG2E  = 1.4426950408889634f;

union Frag16 { v16bf v; uint4 q[2]; };

// Load a 16x32 (A) or 32x16-as-rows (B) bf16 fragment from row-major storage.
// Per ISA 7.12.2: lanes 0-15 hold K {0..7,16..23}, lanes 16-31 hold K {8..15,24..31}.
__device__ __forceinline__ v16bf load_frag_bf16(const __bf16* __restrict__ base,
                                                int row16, int ld, int k0) {
    int lane = threadIdx.x & 31;
    int hh   = lane >> 4;
    int l16  = lane & 15;
    const __bf16* p = base + (size_t)(row16 + l16) * ld + (k0 + hh * 8);
    Frag16 f;
    f.q[0] = *(const uint4*)p;
    f.q[1] = *(const uint4*)(p + 16);
    return f.v;
}

__device__ __forceinline__ v8f wmma_bf16(v16bf a, v16bf b, v8f c) {
    return __builtin_amdgcn_wmma_f32_16x16x32_bf16(false, a, false, b,
                                                   (short)0, c, false, false);
}

// ---------------------------------------------------------------- converts
__global__ __launch_bounds__(256) void cvt_x_kernel(const float* __restrict__ x,
                                                    __bf16* __restrict__ Xbf, int n4) {
    int t = blockIdx.x * 256 + threadIdx.x;
    if (t >= n4) return;
    float4 v = ((const float4*)x)[t];
    union { __bf16 h[4]; uint2 u; } pk;
    pk.h[0] = (__bf16)v.x; pk.h[1] = (__bf16)v.y;
    pk.h[2] = (__bf16)v.z; pk.h[3] = (__bf16)v.w;
    ((uint2*)Xbf)[t] = pk.u;
}

// W[g][i][j] (d_in x d_out) -> Wt[g][j][i] (d_out x d_in), bf16
__global__ __launch_bounds__(256) void cvt_w_kernel(const float* __restrict__ w,
                                                    __bf16* __restrict__ Wt) {
    int t = blockIdx.x * 256 + threadIdx.x;       // 3*1024*1024 threads
    int g = t >> 20, rem = t & 1048575;
    int i = rem >> 10, j = rem & 1023;
    Wt[(size_t)g * (D * D) + (size_t)j * D + i] = (__bf16)w[t];
}

// ---------------------------------------------------------------- QKV GEMM
// grid = (BS/256, D/64, 3), block = 256 (8 waves). Wave -> 32x64 tile:
// each B (weight) fragment is shared by two A fragments (8 WMMAs per 6 loads).
// k-loop unrolled x2 with two fragment buffer sets that alternate roles, so
// the software pipeline needs no register copies.
__global__ __launch_bounds__(256) void qkv_kernel(const __bf16* __restrict__ Xbf,
                                                  const __bf16* __restrict__ Wt,
                                                  __bf16* __restrict__ Qb,
                                                  __bf16* __restrict__ Kb,
                                                  __bf16* __restrict__ Vt) {
    int wave = threadIdx.x >> 5;
    int lane = threadIdx.x & 31;
    int hh   = lane >> 4;
    int l16  = lane & 15;
    int g    = blockIdx.z;
    int m0   = blockIdx.x * 256 + wave * 32;      // row in [0, B*S)
    int n0   = blockIdx.y * 64;                   // out-column base
    const __bf16* W = Wt + (size_t)g * D * D;

    v8f acc[8] = {};                               // acc[mt*4+nt]
    v16bf a0[2], b0[4];
#pragma unroll
    for (int mt = 0; mt < 2; ++mt) a0[mt] = load_frag_bf16(Xbf, m0 + mt * 16, D, 0);
#pragma unroll
    for (int nt = 0; nt < 4; ++nt) b0[nt] = load_frag_bf16(W, n0 + nt * 16, D, 0);

    for (int k0 = 0; k0 < D; k0 += 64) {
        int ka = k0 + 32;                          // always < D (D multiple of 64)
        // load buffer set 1 (k0+32), then consume buffer set 0
        v16bf a1[2], b1[4];
#pragma unroll
        for (int mt = 0; mt < 2; ++mt) a1[mt] = load_frag_bf16(Xbf, m0 + mt * 16, D, ka);
#pragma unroll
        for (int nt = 0; nt < 4; ++nt) b1[nt] = load_frag_bf16(W, n0 + nt * 16, D, ka);
        {   // stream-ahead prefetch (global_prefetch_b8) two k-blocks out
            int kp = (k0 + 128 < D) ? k0 + 128 : k0;
            __builtin_prefetch(Xbf + (size_t)(m0 + lane) * D + kp, 0, 3);
            __builtin_prefetch(W + (size_t)(n0 + (lane & 63)) * D + kp, 0, 3);
        }
#pragma unroll
        for (int mt = 0; mt < 2; ++mt)
#pragma unroll
            for (int nt = 0; nt < 4; ++nt)
                acc[mt * 4 + nt] = wmma_bf16(a0[mt], b0[nt], acc[mt * 4 + nt]);

        // reload buffer set 0 (k0+64, clamped on last trip), consume set 1
        int kb = (k0 + 64 < D) ? k0 + 64 : 0;
#pragma unroll
        for (int mt = 0; mt < 2; ++mt) a0[mt] = load_frag_bf16(Xbf, m0 + mt * 16, D, kb);
#pragma unroll
        for (int nt = 0; nt < 4; ++nt) b0[nt] = load_frag_bf16(W, n0 + nt * 16, D, kb);
#pragma unroll
        for (int mt = 0; mt < 2; ++mt)
#pragma unroll
            for (int nt = 0; nt < 4; ++nt)
                acc[mt * 4 + nt] = wmma_bf16(a1[mt], b1[nt], acc[mt * 4 + nt]);
    }

    if (g == 2) {
        // V stored transposed: Vt[batch][out_col][s]; lane's 8 C values are
        // consecutive M => one contiguous 16B store per lane per tile.
        int batch = m0 / S, sr = m0 % S;
        __bf16* vt = Vt + (size_t)batch * D * S;
#pragma unroll
        for (int mt = 0; mt < 2; ++mt)
#pragma unroll
            for (int nt = 0; nt < 4; ++nt) {
                union { __bf16 h[8]; uint4 q; } pk;
#pragma unroll
                for (int r = 0; r < 8; ++r) pk.h[r] = (__bf16)acc[mt * 4 + nt][r];
                *(uint4*)(vt + (size_t)(n0 + nt * 16 + l16) * S +
                          sr + mt * 16 + hh * 8) = pk.q;
            }
    } else {
        __bf16* dst = (g == 0 ? Qb : Kb);
#pragma unroll
        for (int mt = 0; mt < 2; ++mt)
#pragma unroll
            for (int nt = 0; nt < 4; ++nt)
#pragma unroll
                for (int r = 0; r < 8; ++r)
                    dst[(size_t)(m0 + mt * 16 + hh * 8 + r) * D +
                        n0 + nt * 16 + l16] = (__bf16)acc[mt * 4 + nt][r];
    }
}

// ---------------------------------------------------------------- attention
// grid = (S/16, B), block = 256 (8 waves), dynamic LDS: 16x2048 f32 scores + stats.
__global__ __launch_bounds__(256) void attn_kernel(const __bf16* __restrict__ Qb,
                                                   const __bf16* __restrict__ Kb,
                                                   const __bf16* __restrict__ Vt,
                                                   float* __restrict__ Out) {
    extern __shared__ float smem[];
    float* Ssm  = smem;                 // [16][2048]
    float* rowM = smem + 16 * 2048;     // [16]
    float* rowL = rowM + 16;            // [16] : 1/sum

    int wave = threadIdx.x >> 5;
    int lane = threadIdx.x & 31;
    int hh   = lane >> 4;
    int l16  = lane & 15;
    int q0   = blockIdx.x * 16;
    int b    = blockIdx.y;
    const __bf16* Qp = Qb + (size_t)b * S * D;
    const __bf16* Kp = Kb + (size_t)b * S * D;
    const __bf16* Vp = Vt + (size_t)b * D * S;

    // ---- Pass 1: S = scale * Q K^T, waves split the 2048 keys (256 each);
    // 64 keys per step: the Q fragment feeds 4 WMMAs per 5 fragment loads.
    for (int kt = 0; kt < 4; ++kt) {
        int key0 = wave * 256 + kt * 64;
        v8f sacc[4] = {};
        v16bf qa0 = load_frag_bf16(Qp, q0, D, 0);
        v16bf kf0[4];
#pragma unroll
        for (int j = 0; j < 4; ++j) kf0[j] = load_frag_bf16(Kp, key0 + j * 16, D, 0);
        for (int d0 = 0; d0 < D; d0 += 64) {
            int da = d0 + 32;
            v16bf qa1 = load_frag_bf16(Qp, q0, D, da);
            v16bf kf1[4];
#pragma unroll
            for (int j = 0; j < 4; ++j)
                kf1[j] = load_frag_bf16(Kp, key0 + j * 16, D, da);
#pragma unroll
            for (int j = 0; j < 4; ++j) sacc[j] = wmma_bf16(qa0, kf0[j], sacc[j]);
            int db = (d0 + 64 < D) ? d0 + 64 : 0;
            qa0 = load_frag_bf16(Qp, q0, D, db);
#pragma unroll
            for (int j = 0; j < 4; ++j)
                kf0[j] = load_frag_bf16(Kp, key0 + j * 16, D, db);
#pragma unroll
            for (int j = 0; j < 4; ++j) sacc[j] = wmma_bf16(qa1, kf1[j], sacc[j]);
        }
#pragma unroll
        for (int j = 0; j < 4; ++j)
#pragma unroll
            for (int r = 0; r < 8; ++r)
                Ssm[(hh * 8 + r) * 2048 + key0 + j * 16 + l16] = sacc[j][r] * SCALE;
    }
    __syncthreads();

    // ---- Pass 2: per-row softmax stats (2 rows per wave), wave32 shuffles
    for (int rr = 0; rr < 2; ++rr) {
        int row = wave * 2 + rr;
        const float* sr = Ssm + row * 2048;
        float mx = -3.402823e38f;
        for (int j = lane; j < 2048; j += 32) mx = fmaxf(mx, sr[j]);
#pragma unroll
        for (int off = 16; off; off >>= 1) mx = fmaxf(mx, __shfl_xor(mx, off, 32));
        float l = 0.f;
        for (int j = lane; j < 2048; j += 32) l += exp2f((sr[j] - mx) * LOG2E);
#pragma unroll
        for (int off = 16; off; off >>= 1) l += __shfl_xor(l, off, 32);
        if (lane == 0) { rowM[row] = mx; rowL[row] = 1.0f / l; }
    }
    __syncthreads();

    // ---- Pass 3: O = softmax(S) * V ; waves split 1024 out-cols (128 each)
    int n0 = wave * 128;
    v8f acc[8] = {};
    float mrow = rowM[l16];             // A-fragment row for this lane
    v16bf vb = load_frag_bf16(Vp, n0, S, 0);      // pipelined V fragment
    for (int k0 = 0; k0 < 2048; k0 += 32) {
        const float* sp = Ssm + l16 * 2048 + k0 + hh * 8;
        v16bf pf;
#pragma unroll
        for (int i = 0; i < 8; ++i) {
            pf[i]     = (__bf16)exp2f((sp[i]      - mrow) * LOG2E);
            pf[i + 8] = (__bf16)exp2f((sp[i + 16] - mrow) * LOG2E);
        }
        int k1 = (k0 + 32 < 2048) ? k0 + 32 : 0;
        __builtin_prefetch(Vp + (size_t)(n0 + wave * 16 + l16) * S + k1, 0, 3);
#pragma unroll
        for (int ct = 0; ct < 8; ++ct) {
            // fetch the next fragment (ct+1, or ct=0 of next k-block) while
            // the current WMMA runs; ct-loop is fully unrolled so these
            // rotate through fresh registers without copies
            int nct = (ct + 1) & 7;
            int nk  = (ct == 7) ? k1 : k0;
            v16bf vn = load_frag_bf16(Vp, n0 + nct * 16, S, nk);
            acc[ct] = wmma_bf16(pf, vb, acc[ct]);
            vb = vn;
        }
    }
    float linv[8];
#pragma unroll
    for (int r = 0; r < 8; ++r) linv[r] = rowL[hh * 8 + r];
    float* outp = Out + (size_t)b * S * D;
#pragma unroll
    for (int ct = 0; ct < 8; ++ct)
#pragma unroll
        for (int r = 0; r < 8; ++r)
            outp[(size_t)(q0 + hh * 8 + r) * D + n0 + ct * 16 + l16] =
                acc[ct][r] * linv[r];
}

// ---------------------------------------------------------------- launcher
extern "C" void kernel_launch(void* const* d_in, const int* in_sizes, int n_in,
                              void* d_out, int out_size, void* d_ws, size_t ws_size,
                              hipStream_t stream) {
    const float* x = (const float*)d_in[0];
    const float* w = (const float*)d_in[1];
    float* out = (float*)d_out;

    char* ws = (char*)d_ws;
    size_t off = 0;
    __bf16* Xbf = (__bf16*)(ws + off); off += (size_t)B * S * D * 2;
    __bf16* Wt  = (__bf16*)(ws + off); off += (size_t)3 * D * D * 2;
    __bf16* Qb  = (__bf16*)(ws + off); off += (size_t)B * S * D * 2;
    __bf16* Kb  = (__bf16*)(ws + off); off += (size_t)B * S * D * 2;
    __bf16* Vt  = (__bf16*)(ws + off); off += (size_t)B * S * D * 2;

    int n4 = B * S * D / 4;
    cvt_x_kernel<<<(n4 + 255) / 256, 256, 0, stream>>>(x, Xbf, n4);
    cvt_w_kernel<<<(3 * D * D) / 256, 256, 0, stream>>>(w, Wt);

    qkv_kernel<<<dim3(B * S / 256, D / 64, 3), 256, 0, stream>>>(Xbf, Wt, Qb, Kb, Vt);

    size_t smem_bytes = (16 * 2048 + 32) * sizeof(float);   // 128 KB + stats
    attn_kernel<<<dim3(S / 16, B), 256, smem_bytes, stream>>>(Qb, Kb, Vt, out);
}
